// EdgePredictorResGCN_36197984370747
// MI455X (gfx1250) — matrix operand order repaired and checked by reading, not compiled, not run on hardware
//
#include <hip/hip_runtime.h>
#include <hip/hip_bf16.h>

#define NNODES 50000
#define NEDGES 800000
#define INCH   128
#define HID    64
#define MLPHID 128

typedef _Float16 v16h_t __attribute__((ext_vector_type(16)));
typedef _Float16 v8h_t  __attribute__((ext_vector_type(8)));
typedef float    v8f_t  __attribute__((ext_vector_type(8)));

// ---------------------------------------------------------------------------
// helpers
// ---------------------------------------------------------------------------
__device__ __forceinline__ v8f_t wmma_f16(v16h_t a, v16h_t b, v8f_t c) {
  return __builtin_amdgcn_wmma_f32_16x16x32_f16(false, a, false, b, (short)0, c,
                                                false, false);
}

__device__ __forceinline__ void atomicAddF32(float* p, float v) {
  unsafeAtomicAdd(p, v);   // native global_atomic_add_f32 (no return)
}

// A-fragment (16x32 f16) built from an f32 row. Lane L: m = L&15.
// lanes<16: K = k0..k0+7 , k0+16..k0+23  with k0 = ktbase
// lanes>=16: same with k0 = ktbase+8
__device__ __forceinline__ v16h_t load_a_frag_f32(const float* __restrict__ row,
                                                  int k0, float scale) {
  const float4* p = (const float4*)(row + k0);
  const float4* q = (const float4*)(row + k0 + 16);
  float4 a0 = p[0], a1 = p[1], b0 = q[0], b1 = q[1];
  v16h_t r;
  r[0]  = (_Float16)(a0.x * scale); r[1]  = (_Float16)(a0.y * scale);
  r[2]  = (_Float16)(a0.z * scale); r[3]  = (_Float16)(a0.w * scale);
  r[4]  = (_Float16)(a1.x * scale); r[5]  = (_Float16)(a1.y * scale);
  r[6]  = (_Float16)(a1.z * scale); r[7]  = (_Float16)(a1.w * scale);
  r[8]  = (_Float16)(b0.x * scale); r[9]  = (_Float16)(b0.y * scale);
  r[10] = (_Float16)(b0.z * scale); r[11] = (_Float16)(b0.w * scale);
  r[12] = (_Float16)(b1.x * scale); r[13] = (_Float16)(b1.y * scale);
  r[14] = (_Float16)(b1.z * scale); r[15] = (_Float16)(b1.w * scale);
  return r;
}

__device__ __forceinline__ v16h_t load_a_frag_f16(const _Float16* row, int k0) {
  v8h_t lo = *(const v8h_t*)(row + k0);
  v8h_t hi = *(const v8h_t*)(row + k0 + 16);
  v16h_t r;
#pragma unroll
  for (int i = 0; i < 8; ++i) { r[i] = lo[i]; r[i + 8] = hi[i]; }
  return r;
}

// ---------------------------------------------------------------------------
// weight pack: f32 [OutDim, InDim] row-major -> f16 B-fragments
// frag f = kt*ntiles + nt ; within frag: lane*16 + e
// value = W[(nt*16 + (lane&15)) * InDim + kt*32 + (lane>=16 ? 8 : 0) + (e&7) + (e>=8 ? 16 : 0)]
// ---------------------------------------------------------------------------
__global__ void pack_weight_kernel(const float* __restrict__ W,
                                   _Float16* __restrict__ P,
                                   int ntiles, int InDim, int total) {
  int idx = blockIdx.x * blockDim.x + threadIdx.x;
  if (idx >= total) return;
  int frag = idx >> 9;
  int within = idx & 511;
  int lane = within >> 4, e = within & 15;
  int kt = frag / ntiles, nt = frag - kt * ntiles;
  int n = nt * 16 + (lane & 15);
  int k = kt * 32 + ((lane >> 4) << 3) + (e & 7) + ((e >> 3) << 4);
  P[idx] = (_Float16)W[n * InDim + k];
}

// ---------------------------------------------------------------------------
// degree count + scatter (mean aggregation numerator)
// ---------------------------------------------------------------------------
__global__ void count_kernel(const int* __restrict__ tgt, float* __restrict__ cnt,
                             int nEdges) {
  int e = blockIdx.x * blockDim.x + threadIdx.x;
  if (e >= nEdges) return;
  atomicAddF32(&cnt[tgt[e]], 1.0f);
}

__global__ void scatter_kernel(const float* __restrict__ x,
                               const int* __restrict__ src,
                               const int* __restrict__ tgt,
                               float* __restrict__ agg, int C4, int nEdges) {
  int idx = blockIdx.x * blockDim.x + threadIdx.x;
  int e = idx / C4;
  if (e >= nEdges) return;
  int c = (idx - e * C4) * 4;
  int C = C4 * 4;
  float4 v = *(const float4*)(x + (size_t)src[e] * C + c);
  float* dst = agg + (size_t)tgt[e] * C + c;
  atomicAddF32(dst + 0, v.x);
  atomicAddF32(dst + 1, v.y);
  atomicAddF32(dst + 2, v.z);
  atomicAddF32(dst + 3, v.w);
}

// ---------------------------------------------------------------------------
// node layer pass 1: pre = mean @ Wl^T + x @ Wr^T + bl ; (res = x @ rW^T + rb)
// also accumulates per-column sum / sumsq for BatchNorm.
// One wave per 16-node tile; KDIM = 128 or 64; out = HID = 64 (4 n-tiles).
// ---------------------------------------------------------------------------
template <int KDIM, bool HAS_RES>
__global__ void __launch_bounds__(256)
node_gemm_kernel(const float* __restrict__ agg, const float* __restrict__ cnt,
                 const float* __restrict__ xin,
                 const _Float16* __restrict__ WlP, const _Float16* __restrict__ WrP,
                 const _Float16* __restrict__ rWP,
                 const float* __restrict__ bl, const float* __restrict__ rb,
                 float* __restrict__ pre, float* __restrict__ resb,
                 float* __restrict__ colsum, float* __restrict__ colsumsq,
                 int nTiles) {
  int wave = threadIdx.x >> 5, lane = threadIdx.x & 31;
  int tile = blockIdx.x * 8 + wave;
  if (tile >= nTiles) return;          // whole-wave exit: EXEC stays all-ones
  int row0 = tile * 16;
  int m = lane & 15;
  int hiK = (lane >> 4) * 8;
  const float* aggRow = agg + (size_t)(row0 + m) * KDIM;
  const float* xRow   = xin + (size_t)(row0 + m) * KDIM;
  float inv = 1.0f / fmaxf(cnt[row0 + m], 1.0f);

  v8f_t accP[4] = {};
  v8f_t accR[4] = {};

  constexpr int KT = KDIM / 32;
#pragma unroll
  for (int kt = 0; kt < KT; ++kt) {
    int k0 = kt * 32 + hiK;
    v16h_t am = load_a_frag_f32(aggRow, k0, inv);
    v16h_t ax = load_a_frag_f32(xRow, k0, 1.0f);
#pragma unroll
    for (int nt = 0; nt < 4; ++nt) {
      v16h_t bL = *(const v16h_t*)(WlP + ((kt * 4 + nt) << 9) + lane * 16);
      accP[nt] = wmma_f16(am, bL, accP[nt]);
      v16h_t bR = *(const v16h_t*)(WrP + ((kt * 4 + nt) << 9) + lane * 16);
      accP[nt] = wmma_f16(ax, bR, accP[nt]);
      if constexpr (HAS_RES) {
        v16h_t bS = *(const v16h_t*)(rWP + ((kt * 4 + nt) << 9) + lane * 16);
        accR[nt] = wmma_f16(ax, bS, accR[nt]);
      }
    }
  }

  int mBase = (lane >> 4) * 8;   // D rows: M = r + mBase, N = (lane&15)+nt*16
#pragma unroll
  for (int nt = 0; nt < 4; ++nt) {
    int n = (lane & 15) + nt * 16;
    float bias = bl[n];
    float rbias = 0.0f;
    if constexpr (HAS_RES) rbias = rb[n];
    float s = 0.0f, ss = 0.0f;
#pragma unroll
    for (int r = 0; r < 8; ++r) {
      float v = accP[nt][r] + bias;
      pre[(size_t)(row0 + mBase + r) * HID + n] = v;
      s += v; ss += v * v;
      if constexpr (HAS_RES)
        resb[(size_t)(row0 + mBase + r) * HID + n] = accR[nt][r] + rbias;
    }
    s  += __shfl_xor(s, 16, 32);
    ss += __shfl_xor(ss, 16, 32);
    if (lane < 16) {
      atomicAddF32(&colsum[n], s);
      atomicAddF32(&colsumsq[n], ss);
    }
  }
}

__global__ void bnstats_kernel(const float* __restrict__ colsum,
                               const float* __restrict__ colsumsq,
                               float* __restrict__ mu, float* __restrict__ rstd,
                               float invN) {
  int c = threadIdx.x;   // 64 threads
  float m = colsum[c] * invN;
  float var = colsumsq[c] * invN - m * m;
  mu[c] = m;
  rstd[c] = rsqrtf(var + 1e-5f);
}

__global__ void bn_apply_kernel(const float* __restrict__ pre,
                                const float* __restrict__ resid,
                                const float* __restrict__ mu,
                                const float* __restrict__ rstd,
                                const float* __restrict__ g,
                                const float* __restrict__ be,
                                float* __restrict__ hout, int total) {
  int idx = blockIdx.x * blockDim.x + threadIdx.x;
  if (idx >= total) return;
  int c = idx & (HID - 1);
  float v = (pre[idx] - mu[c]) * rstd[c] * g[c] + be[c];
  hout[idx] = fmaxf(v, 0.0f) + resid[idx];
}

__global__ void f32_to_f16_kernel(const float* __restrict__ in,
                                  _Float16* __restrict__ out, int total) {
  int idx = blockIdx.x * blockDim.x + threadIdx.x;
  if (idx >= total) return;
  out[idx] = (_Float16)in[idx];
}

// ---------------------------------------------------------------------------
// edge MLP: f = [h[src] | h[tgt]] (128) -> relu(W1 f + b1) -> relu(W2 z + b2)
//           -> W3 . z2 + b3
// One wave per 16-edge tile; z1 staged in padded LDS (wave-private).
// ---------------------------------------------------------------------------
__global__ void __launch_bounds__(128)
edge_mlp_kernel(const _Float16* __restrict__ hh, const int* __restrict__ src,
                const int* __restrict__ tgt,
                const _Float16* __restrict__ W1P, const float* __restrict__ b1,
                const _Float16* __restrict__ W2P, const float* __restrict__ b2,
                const float* __restrict__ W3, const float* __restrict__ b3,
                float* __restrict__ out) {
  __shared__ __align__(16) _Float16 stage[4][16][136];   // 136 pad: rotate banks
  int wave = threadIdx.x >> 5, lane = threadIdx.x & 31;
  int e0 = (blockIdx.x * 4 + wave) * 16;
  int m = lane & 15;
  int hiK = (lane >> 4) * 8;
  int e = e0 + m;
  const _Float16* hs = hh + (size_t)src[e] * HID;
  const _Float16* ht = hh + (size_t)tgt[e] * HID;

  // --- layer 1: z1 = relu(f @ W1^T + b1), 8 n-tiles x 4 k-tiles ---
  v8f_t acc[8] = {};
#pragma unroll
  for (int kt = 0; kt < 4; ++kt) {
    const _Float16* rp = (kt < 2) ? hs : ht;
    int k0 = (kt & 1) * 32 + hiK;
    v16h_t a = load_a_frag_f16(rp, k0);
#pragma unroll
    for (int nt = 0; nt < 8; ++nt) {
      v16h_t b = *(const v16h_t*)(W1P + ((kt * 8 + nt) << 9) + lane * 16);
      acc[nt] = wmma_f16(a, b, acc[nt]);
    }
  }
  int mBase = (lane >> 4) * 8;
#pragma unroll
  for (int nt = 0; nt < 8; ++nt) {
    int n = m + nt * 16;
    float bias = b1[n];
#pragma unroll
    for (int r = 0; r < 8; ++r) {
      float v = fmaxf(acc[nt][r] + bias, 0.0f);
      stage[wave][mBase + r][n] = (_Float16)v;
    }
  }
  asm volatile("s_wait_dscnt 0" ::: "memory");  // wave-local LDS RAW fence

  // --- layer 2: z2 = relu(z1 @ W2^T + b2) ---
  v8f_t acc2[8] = {};
  const _Float16* srow = &stage[wave][m][0];
#pragma unroll
  for (int kt = 0; kt < 4; ++kt) {
    int k0 = kt * 32 + hiK;
    v16h_t a = load_a_frag_f16(srow, k0);
#pragma unroll
    for (int nt = 0; nt < 8; ++nt) {
      v16h_t b = *(const v16h_t*)(W2P + ((kt * 8 + nt) << 9) + lane * 16);
      acc2[nt] = wmma_f16(a, b, acc2[nt]);
    }
  }

  // --- output layer: dot with W3 (N=1 -> VALU + shuffle reduce) ---
  float psum[8] = {0, 0, 0, 0, 0, 0, 0, 0};
#pragma unroll
  for (int nt = 0; nt < 8; ++nt) {
    int n = m + nt * 16;
    float bias = b2[n];
    float w3 = W3[n];
#pragma unroll
    for (int r = 0; r < 8; ++r)
      psum[r] += fmaxf(acc2[nt][r] + bias, 0.0f) * w3;
  }
#pragma unroll
  for (int mask = 1; mask <= 8; mask <<= 1)
#pragma unroll
    for (int r = 0; r < 8; ++r)
      psum[r] += __shfl_xor(psum[r], mask, 32);
  if ((lane & 15) == 0) {
    float b3v = b3[0];
#pragma unroll
    for (int r = 0; r < 8; ++r)
      out[e0 + mBase + r] = psum[r] + b3v;
  }
}

// ---------------------------------------------------------------------------
// host
// ---------------------------------------------------------------------------
static inline size_t alnup(size_t x) { return (x + 255) & ~(size_t)255; }

extern "C" void kernel_launch(void* const* d_in, const int* in_sizes, int n_in,
                              void* d_out, int out_size, void* d_ws, size_t ws_size,
                              hipStream_t stream) {
  const float* x   = (const float*)d_in[0];
  const int* ei    = (const int*)d_in[1];
  const int* src   = ei;
  const int* tgt   = ei + NEDGES;
  const float* Wl0 = (const float*)d_in[2];
  const float* bl0 = (const float*)d_in[3];
  const float* Wr0 = (const float*)d_in[4];
  const float* g0  = (const float*)d_in[5];
  const float* be0 = (const float*)d_in[6];
  const float* rW0 = (const float*)d_in[7];
  const float* rb0 = (const float*)d_in[8];
  const float* Wl1 = (const float*)d_in[9];
  const float* bl1 = (const float*)d_in[10];
  const float* Wr1 = (const float*)d_in[11];
  const float* g1  = (const float*)d_in[12];
  const float* be1 = (const float*)d_in[13];
  const float* Wl2 = (const float*)d_in[14];
  const float* bl2 = (const float*)d_in[15];
  const float* Wr2 = (const float*)d_in[16];
  const float* g2  = (const float*)d_in[17];
  const float* be2 = (const float*)d_in[18];
  const float* W1  = (const float*)d_in[19];
  const float* b1  = (const float*)d_in[20];
  const float* W2  = (const float*)d_in[21];
  const float* b2  = (const float*)d_in[22];
  const float* W3  = (const float*)d_in[23];
  const float* b3  = (const float*)d_in[24];
  float* out = (float*)d_out;

  uint8_t* ws = (uint8_t*)d_ws;
  size_t off = 0;
  float* cnt = (float*)(ws + off);       off = alnup(off + NNODES * 4);
  float* agg = (float*)(ws + off);       off = alnup(off + (size_t)NNODES * INCH * 4);
  float* pre = (float*)(ws + off);       off = alnup(off + (size_t)NNODES * HID * 4);
  float* resb = (float*)(ws + off);      off = alnup(off + (size_t)NNODES * HID * 4);
  float* hA = (float*)(ws + off);        off = alnup(off + (size_t)NNODES * HID * 4);
  float* hB = (float*)(ws + off);        off = alnup(off + (size_t)NNODES * HID * 4);
  float* stats = (float*)(ws + off);     off = alnup(off + 256 * 4);
  float* colsum = stats, *colsumsq = stats + 64, *mu = stats + 128, *rstd = stats + 192;
  _Float16* Wl0P = (_Float16*)(ws + off); off = alnup(off + HID * INCH * 2);
  _Float16* Wr0P = (_Float16*)(ws + off); off = alnup(off + HID * INCH * 2);
  _Float16* rW0P = (_Float16*)(ws + off); off = alnup(off + HID * INCH * 2);
  _Float16* Wl1P = (_Float16*)(ws + off); off = alnup(off + HID * HID * 2);
  _Float16* Wr1P = (_Float16*)(ws + off); off = alnup(off + HID * HID * 2);
  _Float16* Wl2P = (_Float16*)(ws + off); off = alnup(off + HID * HID * 2);
  _Float16* Wr2P = (_Float16*)(ws + off); off = alnup(off + HID * HID * 2);
  _Float16* W1P  = (_Float16*)(ws + off); off = alnup(off + MLPHID * 2 * HID * 2);
  _Float16* W2P  = (_Float16*)(ws + off); off = alnup(off + MLPHID * MLPHID * 2);
  _Float16* hh = (_Float16*)agg;  // alias: agg dead before hh is written

  // --- pack weights to f16 B-fragments ---
  pack_weight_kernel<<<(HID * INCH + 255) / 256, 256, 0, stream>>>(Wl0, Wl0P, 4, INCH, HID * INCH);
  pack_weight_kernel<<<(HID * INCH + 255) / 256, 256, 0, stream>>>(Wr0, Wr0P, 4, INCH, HID * INCH);
  pack_weight_kernel<<<(HID * INCH + 255) / 256, 256, 0, stream>>>(rW0, rW0P, 4, INCH, HID * INCH);
  pack_weight_kernel<<<(HID * HID + 255) / 256, 256, 0, stream>>>(Wl1, Wl1P, 4, HID, HID * HID);
  pack_weight_kernel<<<(HID * HID + 255) / 256, 256, 0, stream>>>(Wr1, Wr1P, 4, HID, HID * HID);
  pack_weight_kernel<<<(HID * HID + 255) / 256, 256, 0, stream>>>(Wl2, Wl2P, 4, HID, HID * HID);
  pack_weight_kernel<<<(HID * HID + 255) / 256, 256, 0, stream>>>(Wr2, Wr2P, 4, HID, HID * HID);
  pack_weight_kernel<<<(MLPHID * 2 * HID + 255) / 256, 256, 0, stream>>>(W1, W1P, 8, 2 * HID, MLPHID * 2 * HID);
  pack_weight_kernel<<<(MLPHID * MLPHID + 255) / 256, 256, 0, stream>>>(W2, W2P, 8, MLPHID, MLPHID * MLPHID);

  // --- degrees (same for all layers) ---
  hipMemsetAsync(cnt, 0, NNODES * 4, stream);
  count_kernel<<<(NEDGES + 255) / 256, 256, 0, stream>>>(tgt, cnt, NEDGES);

  const int nTiles = NNODES / 16;          // 3125
  const int gemmGrid = (nTiles + 7) / 8;
  const float invN = 1.0f / (float)NNODES;

  // --- layer 0 (in 128 -> 64, linear residual) ---
  hipMemsetAsync(agg, 0, (size_t)NNODES * INCH * 4, stream);
  scatter_kernel<<<((size_t)NEDGES * (INCH / 4) + 255) / 256, 256, 0, stream>>>(
      x, src, tgt, agg, INCH / 4, NEDGES);
  hipMemsetAsync(stats, 0, 128 * 4, stream);
  node_gemm_kernel<INCH, true><<<gemmGrid, 256, 0, stream>>>(
      agg, cnt, x, Wl0P, Wr0P, rW0P, bl0, rb0, pre, resb, colsum, colsumsq, nTiles);
  bnstats_kernel<<<1, 64, 0, stream>>>(colsum, colsumsq, mu, rstd, invN);
  bn_apply_kernel<<<(NNODES * HID + 255) / 256, 256, 0, stream>>>(
      pre, resb, mu, rstd, g0, be0, hA, NNODES * HID);

  // --- layer 1 (64 -> 64, identity residual) ---
  hipMemsetAsync(agg, 0, (size_t)NNODES * HID * 4, stream);
  scatter_kernel<<<((size_t)NEDGES * (HID / 4) + 255) / 256, 256, 0, stream>>>(
      hA, src, tgt, agg, HID / 4, NEDGES);
  hipMemsetAsync(stats, 0, 128 * 4, stream);
  node_gemm_kernel<HID, false><<<gemmGrid, 256, 0, stream>>>(
      agg, cnt, hA, Wl1P, Wr1P, nullptr, bl1, nullptr, pre, nullptr, colsum, colsumsq, nTiles);
  bnstats_kernel<<<1, 64, 0, stream>>>(colsum, colsumsq, mu, rstd, invN);
  bn_apply_kernel<<<(NNODES * HID + 255) / 256, 256, 0, stream>>>(
      pre, hA, mu, rstd, g1, be1, hB, NNODES * HID);

  // --- layer 2 (64 -> 64, identity residual) ---
  hipMemsetAsync(agg, 0, (size_t)NNODES * HID * 4, stream);
  scatter_kernel<<<((size_t)NEDGES * (HID / 4) + 255) / 256, 256, 0, stream>>>(
      hB, src, tgt, agg, HID / 4, NEDGES);
  hipMemsetAsync(stats, 0, 128 * 4, stream);
  node_gemm_kernel<HID, false><<<gemmGrid, 256, 0, stream>>>(
      agg, cnt, hB, Wl2P, Wr2P, nullptr, bl2, nullptr, pre, nullptr, colsum, colsumsq, nTiles);
  bnstats_kernel<<<1, 64, 0, stream>>>(colsum, colsumsq, mu, rstd, invN);
  bn_apply_kernel<<<(NNODES * HID + 255) / 256, 256, 0, stream>>>(
      pre, hB, mu, rstd, g2, be2, hA, NNODES * HID);

  // --- final node features -> f16 (aliased over agg) ---
  f32_to_f16_kernel<<<(NNODES * HID + 255) / 256, 256, 0, stream>>>(hA, hh, NNODES * HID);

  // --- edge MLP (compute-dominant stage: 64 WMMAs / 16 edges) ---
  edge_mlp_kernel<<<NEDGES / 64, 128, 0, stream>>>(
      hh, src, tgt, W1P, b1, W2P, b2, W3, b3, out);

  (void)in_sizes; (void)n_in; (void)out_size; (void)ws_size;
}
